// MemorySaverIPRMPNNModel_89876485636294
// MI455X (gfx1250) — matrix operand order repaired and checked by reading, not compiled, not run on hardware
//
#include <hip/hip_runtime.h>
#include <hip/hip_bf16.h>
#include <math.h>

// ---------------------------------------------------------------------------
// MemorySaverIPRMPNN fused pipeline for MI455X (gfx1250, wave32, WMMA).
// One workgroup = 8 graphs = 160 nodes; all intermediates live in WGP LDS.
// Weights are pre-swizzled into WMMA-B fragment-major layout so each B
// fragment is one aligned 32-byte load; each wave owns one 16-col tile and
// keeps its B fragments in registers across all 10 row tiles.
// ---------------------------------------------------------------------------

typedef __attribute__((ext_vector_type(16))) _Float16 v16h;
typedef __attribute__((ext_vector_type(8)))  float    v8f;

#define NP      20
#define DEG     4
#define DIN     64
#define HDIM    128
#define DOUT    10
#define GPB     8                   // graphs per block
#define ROWS    (GPB * NP)          // 160 node-rows per block
#define EPB     (GPB * NP * DEG)    // 640 edges per block
#define NTHREADS 256
#define NWAVES   8
#define RTILES   (ROWS / 16)        // 10
#define CTILES   (HDIM / 16)        // 8

// f16 weight blob layout inside d_ws (elements), fragment-major packed
#define W16_EMB_OFF  0              // 64*128
#define W16_GCN_OFF  8192           // 128*128
#define W16_AFF_OFF  24576          // 128*128
#define W16_TOTAL    40960

// dynamic LDS layout (bytes)
#define OFF_XS      0               // _Float16 [160][64]   (20480 B) -- dead after GEMM1
#define OFF_H       20480           // _Float16 [160][128]  (40960 B) -- dead after GEMM2
#define OFF_HW      61440           // float    [160][128]  (81920 B)
#define OFF_AGG     143360          // float    [160][128]  (81920 B)
#define OFF_COEF    225280          // float    [160]
#define OFF_ROWSUM  225920          // float    [160]
#define OFF_FSCALE  226560          // float    [160]
#define OFF_S       227200          // float    [8][128]    (4096 B)
#define OFF_EDGE    231296          // int      [640][2]    (5120 B)
#define OFF_HX      0               // _Float16 [160][128]  aliases XS+H (40960 B)
#define LDS_BYTES   236416

// ---------------- WMMA helpers (CDNA5 16x16x32 f16 -> f32) ------------------

__device__ __forceinline__ v8f wmma_f16(v16h a, v16h b, v8f c) {
  // (neg_a, A, neg_b, B, c_mod, C, reuse_a, reuse_b)
  return __builtin_amdgcn_wmma_f32_16x16x32_f16(false, a, false, b, (short)0, c,
                                                false, false);
}

// A fragment 16x32 (MxK): lane<16 holds row (lane&15), K in {0..7}u{16..23};
// lane>=16 holds same row, K in {8..15}u{24..31}.
__device__ __forceinline__ v16h load_a_frag(const _Float16* base, int lda,
                                            int row0, int k0, int lane) {
  const _Float16* p = base + (row0 + (lane & 15)) * lda + k0 + ((lane >> 4) << 3);
  v16h a;
#pragma unroll
  for (int i = 0; i < 8; ++i) { a[i] = p[i]; a[8 + i] = p[16 + i]; }
  return a;
}

// Pre-swizzled B fragment: one aligned 32B vector per (ct, kt, lane).
__device__ __forceinline__ v16h load_b_packed(const _Float16* base, int Kdim,
                                              int ct, int kt, int lane) {
  return *(const v16h*)(base + ((((ct * (Kdim / 32)) + kt) * 32 + lane) << 4));
}

// C tile: lane holds col (lane&15); acc[j] is row row0 + j + 8*(lane>>4).
__device__ __forceinline__ void store_c_f32(float* base, int ldc, int row0,
                                            int col0, int lane, v8f acc) {
  int c  = col0 + (lane & 15);
  int r0 = row0 + ((lane >> 4) << 3);
#pragma unroll
  for (int j = 0; j < 8; ++j) base[(r0 + j) * ldc + c] = acc[j];
}

__device__ __forceinline__ void store_c_f16_bias(_Float16* base, int ldc,
                                                 int row0, int col0, int lane,
                                                 v8f acc, float bias) {
  int c  = col0 + (lane & 15);
  int r0 = row0 + ((lane >> 4) << 3);
#pragma unroll
  for (int j = 0; j < 8; ++j)
    base[(r0 + j) * ldc + c] = (_Float16)(acc[j] + bias);
}

// ---------------- weight fp32 -> fp16 fragment-major pre-pack ---------------
// Packed index e (within one weight blob of K rows x 128 cols):
//   ct = e / ((K/32)*512); kt = ...; lane = ...; i = e & 15
//   source element = W[kt*32 + 2*(i&8) + 8*(lane>>4) + (i&7)][ct*16 + (lane&15)]

__device__ __forceinline__ int bsrc_index(int e, int Kdim) {
  const int perct = (Kdim / 32) * 512;
  const int ct   = e / perct;
  const int rem  = e - ct * perct;
  const int kt   = rem >> 9;
  const int r2   = rem & 511;
  const int lane = r2 >> 4;
  const int i    = r2 & 15;
  const int k = kt * 32 + ((i & 8) << 1) + ((lane >> 4) << 3) + (i & 7);
  const int n = ct * 16 + (lane & 15);
  return k * HDIM + n;
}

__global__ void __launch_bounds__(256) pack_weights_kernel(
    const float* __restrict__ Wemb, const float* __restrict__ Wgcn,
    const float* __restrict__ Waff, _Float16* __restrict__ w16) {
  int i = blockIdx.x * blockDim.x + threadIdx.x;
  if (i < W16_GCN_OFF) {
    w16[i] = (_Float16)Wemb[bsrc_index(i, DIN)];
  } else if (i < W16_AFF_OFF) {
    w16[i] = (_Float16)Wgcn[bsrc_index(i - W16_GCN_OFF, HDIM)];
  } else if (i < W16_TOTAL) {
    w16[i] = (_Float16)Waff[bsrc_index(i - W16_AFF_OFF, HDIM)];
  }
}

// ---------------- fused per-8-graph mega kernel ------------------------------

__global__ void __launch_bounds__(NTHREADS, 1) iprmpnn_fused_kernel(
    const float* __restrict__ x,
    const int*   __restrict__ edge_src,
    const int*   __restrict__ edge_dst,
    const float* __restrict__ b_emb,
    const float* __restrict__ b_gcn,
    const float* __restrict__ b_aff,
    const float* __restrict__ W_mlp,
    const float* __restrict__ b_mlp,
    const _Float16* __restrict__ w16,
    float* __restrict__ out) {
  extern __shared__ char smem[];
  _Float16* xs     = (_Float16*)(smem + OFF_XS);
  _Float16* hbuf   = (_Float16*)(smem + OFF_H);
  float*    hw     = (float*)(smem + OFF_HW);
  float*    agg    = (float*)(smem + OFF_AGG);
  float*    coef   = (float*)(smem + OFF_COEF);
  float*    rowsum = (float*)(smem + OFF_ROWSUM);
  float*    fscale = (float*)(smem + OFF_FSCALE);
  float*    S      = (float*)(smem + OFF_S);
  int*      eloc   = (int*)(smem + OFF_EDGE);
  _Float16* hx     = (_Float16*)(smem + OFF_HX);

  const int tid  = (int)threadIdx.x;
  const int lane = tid & 31;
  // force scalar: each wave uniformly owns one 16-column tile
  const int ct   = __builtin_amdgcn_readfirstlane(tid >> 5);
  const int col0 = ct * 16;

  const int gbase    = (int)blockIdx.x * GPB;
  const int nodeBase = gbase * NP;
  const int edgeBase = gbase * NP * DEG;

  // ---- Stage 0: load x (fp32 -> fp16 LDS), localize edges, init scalars ----
  for (int i = tid; i < ROWS * DIN; i += NTHREADS)
    xs[i] = (_Float16)x[(size_t)nodeBase * DIN + i];
  for (int e = tid; e < EPB; e += NTHREADS) {
    eloc[2 * e]     = edge_src[edgeBase + e] - nodeBase;
    eloc[2 * e + 1] = edge_dst[edgeBase + e] - nodeBase;
  }
  for (int i = tid; i < ROWS; i += NTHREADS) {
    coef[i]   = 1.0f;   // self-loop contribution of degree
    rowsum[i] = 0.0f;
  }
  __syncthreads();

  // degree via LDS float atomics, then coef = rsqrt(deg)
  for (int e = tid; e < EPB; e += NTHREADS)
    atomicAdd(&coef[eloc[2 * e + 1]], 1.0f);
  __syncthreads();
  for (int i = tid; i < ROWS; i += NTHREADS) coef[i] = rsqrtf(coef[i]);

  // ---- GEMM1: h = xs @ W_emb + b_emb    (K = 64; B resident in regs) ----
  {
    const _Float16* Wp = w16 + W16_EMB_OFF;
    const v16h b0 = load_b_packed(Wp, DIN, ct, 0, lane);
    const v16h b1 = load_b_packed(Wp, DIN, ct, 1, lane);
    const float bias = b_emb[col0 + (lane & 15)];
    for (int rt = 0; rt < RTILES; ++rt) {
      const int row0 = rt * 16;
      v8f acc = {};
      acc = wmma_f16(load_a_frag(xs, DIN, row0, 0,  lane), b0, acc);
      acc = wmma_f16(load_a_frag(xs, DIN, row0, 32, lane), b1, acc);
      store_c_f16_bias(hbuf, HDIM, row0, col0, lane, acc, bias);
    }
  }
  __syncthreads();

  // ---- GEMM2: hw = h @ W_gcn            (K = 128, fp32 out) ----
  {
    const _Float16* Wp = w16 + W16_GCN_OFF;
    const v16h b0 = load_b_packed(Wp, HDIM, ct, 0, lane);
    const v16h b1 = load_b_packed(Wp, HDIM, ct, 1, lane);
    const v16h b2 = load_b_packed(Wp, HDIM, ct, 2, lane);
    const v16h b3 = load_b_packed(Wp, HDIM, ct, 3, lane);
    for (int rt = 0; rt < RTILES; ++rt) {
      const int row0 = rt * 16;
      v8f acc = {};
      acc = wmma_f16(load_a_frag(hbuf, HDIM, row0, 0,  lane), b0, acc);
      acc = wmma_f16(load_a_frag(hbuf, HDIM, row0, 32, lane), b1, acc);
      acc = wmma_f16(load_a_frag(hbuf, HDIM, row0, 64, lane), b2, acc);
      acc = wmma_f16(load_a_frag(hbuf, HDIM, row0, 96, lane), b3, acc);
      store_c_f32(hw, HDIM, row0, col0, lane, acc);
    }
  }
  for (int i = tid; i < ROWS * HDIM; i += NTHREADS) agg[i] = 0.0f;
  __syncthreads();

  // ---- Edge aggregation: agg[dst] += hw[src] * coef[src] ----
  for (int idx = tid; idx < EPB * HDIM; idx += NTHREADS) {
    const int e  = idx >> 7;
    const int hd = idx & (HDIM - 1);
    const int s  = eloc[2 * e];
    const int d  = eloc[2 * e + 1];
    atomicAdd(&agg[d * HDIM + hd], hw[s * HDIM + hd] * coef[s]);
  }
  __syncthreads();

  // ---- hx = relu(coef*(agg + coef*hw) + b_gcn)  (fp16, aliases xs/h) ----
  for (int i = tid; i < ROWS * HDIM; i += NTHREADS) {
    const int n  = i >> 7;
    const int hd = i & (HDIM - 1);
    const float c = coef[n];
    const float v = c * (agg[i] + c * hw[i]) + b_gcn[hd];
    hx[i] = (_Float16)fmaxf(v, 0.0f);
  }
  __syncthreads();

  // ---- GEMM3: nf = relu(hx @ W_aff + b_aff); accumulate row sums ----
  {
    const _Float16* Wp = w16 + W16_AFF_OFF;
    const v16h b0 = load_b_packed(Wp, HDIM, ct, 0, lane);
    const v16h b1 = load_b_packed(Wp, HDIM, ct, 1, lane);
    const v16h b2 = load_b_packed(Wp, HDIM, ct, 2, lane);
    const v16h b3 = load_b_packed(Wp, HDIM, ct, 3, lane);
    const float bias = b_aff[col0 + (lane & 15)];
    for (int rt = 0; rt < RTILES; ++rt) {
      const int row0 = rt * 16;
      v8f acc = {};
      acc = wmma_f16(load_a_frag(hx, HDIM, row0, 0,  lane), b0, acc);
      acc = wmma_f16(load_a_frag(hx, HDIM, row0, 32, lane), b1, acc);
      acc = wmma_f16(load_a_frag(hx, HDIM, row0, 64, lane), b2, acc);
      acc = wmma_f16(load_a_frag(hx, HDIM, row0, 96, lane), b3, acc);
      const int r0 = row0 + ((lane >> 4) << 3);
#pragma unroll
      for (int j = 0; j < 8; ++j)
        atomicAdd(&rowsum[r0 + j], fmaxf(acc[j] + bias, 0.0f));
    }
  }
  __syncthreads();

  // ---- attention -> exact float chain of template + double top-k prune ----
  for (int n = tid; n < ROWS; n += NTHREADS) {
    const float att = 1.0f / (1.0f + expf(-rowsum[n] * (1.0f / 128.0f)));
    const float tw  = 1.0f / 10.0f;              // uniform template entry
    const float trs = (tw + tw) + tw;            // row sum over top-3 kept
    const float tpl = tw / trs;                  // template value (~1/3)
    const float mwe = tpl * att;
    float rs1 = (mwe + mwe) + mwe; if (rs1 == 0.0f) rs1 = 1.0f;
    const float m = mwe / rs1;
    float rs2 = (m + m) + m;       if (rs2 == 0.0f) rs2 = 1.0f;
    fscale[n] = m / rs2;                         // per-node pooled weight
  }
  __syncthreads();

  // ---- pooled S[g][h] = sum_n fscale * hx; gf = (S+S+S)/10 ----
  for (int i = tid; i < GPB * HDIM; i += NTHREADS) {
    const int g  = i >> 7;
    const int hd = i & (HDIM - 1);
    const _Float16* hp = hx + (g * NP) * HDIM + hd;
    const float*    fp = fscale + g * NP;
    float s = 0.0f;
#pragma unroll
    for (int n = 0; n < NP; ++n) s += fp[n] * (float)hp[n * HDIM];
    S[i] = ((s + s) + s) * 0.1f;   // 3 identical virtual rows, mean over V=10
  }
  __syncthreads();

  // ---- final MLP: out[g] = gf @ W_mlp + b_mlp ----
  for (int t = tid; t < GPB * DOUT; t += NTHREADS) {
    const int g = t / DOUT, o = t - g * DOUT;
    const float* sp = S + g * HDIM;
    float acc = b_mlp[o];
#pragma unroll 4
    for (int k = 0; k < HDIM; ++k) acc += sp[k] * W_mlp[k * DOUT + o];
    out[(size_t)(gbase + g) * DOUT + o] = acc;
  }
}

// ---------------------------------------------------------------------------

extern "C" void kernel_launch(void* const* d_in, const int* in_sizes, int n_in,
                              void* d_out, int out_size, void* d_ws, size_t ws_size,
                              hipStream_t stream) {
  (void)n_in; (void)out_size; (void)ws_size;
  const float* x        = (const float*)d_in[0];
  const int*   edge_src = (const int*)d_in[1];
  const int*   edge_dst = (const int*)d_in[2];
  const float* W_emb    = (const float*)d_in[3];
  const float* b_emb    = (const float*)d_in[4];
  const float* W_gcn    = (const float*)d_in[5];
  const float* b_gcn    = (const float*)d_in[6];
  const float* W_aff    = (const float*)d_in[7];
  const float* b_aff    = (const float*)d_in[8];
  const float* W_mlp    = (const float*)d_in[9];
  const float* b_mlp    = (const float*)d_in[10];
  float*       out      = (float*)d_out;
  _Float16*    w16      = (_Float16*)d_ws;

  // per-launch fp32 -> fp16 weight pack into WMMA-B fragment-major layout
  pack_weights_kernel<<<(W16_TOTAL + 255) / 256, 256, 0, stream>>>(
      W_emb, W_gcn, W_aff, w16);

  const int N = in_sizes[0] / DIN;          // total nodes
  const int numBlocks = N / ROWS;           // 8 graphs (160 nodes) per block
  iprmpnn_fused_kernel<<<numBlocks, NTHREADS, LDS_BYTES, stream>>>(
      x, edge_src, edge_dst, b_emb, b_gcn, b_aff, W_mlp, b_mlp, w16, out);
}
